// SampledSoftmax_7876970021286
// MI455X (gfx1250) — matrix operand order repaired and checked by reading, not compiled, not run on hardware
//
#include <hip/hip_runtime.h>

// ---------------------------------------------------------------------------
// Sampled softmax logits for MI455X (gfx1250, wave32, WMMA).
// B=4096, D=256, V=793470, S=8192. Output: [B, 1+S] logits then B int32 zeros.
//
// Strategy: bf16x3 split (hi+lo) on v_wmma_f32_16x16x32_bf16 for near-f32
// accuracy at matrix-core rates. Two pre-pass kernels convert inputs and the
// gathered sample weights ONCE into hi/lo bf16 stored in the exact per-lane
// WMMA fragment layout (A 16x32 and B 32x16, CDNA5 ISA 7.12.2). The GEMM then
// streams fragments straight from L2 with global_load_b128 -> v_wmma, with no
// LDS staging, no barriers, and no per-block conversion VALU.
// ---------------------------------------------------------------------------

typedef __attribute__((ext_vector_type(8)))  float          v8f;
typedef __attribute__((ext_vector_type(8)))  __bf16         v8bf;
typedef __attribute__((ext_vector_type(16))) __bf16         v16bf;
typedef __attribute__((ext_vector_type(8)))  unsigned short v8us;

#define BDIM 4096
#define DDIM 256
#define SDIM 8192
#define OUTSTRIDE (SDIM + 1)
#define MT (BDIM / 16)   // 256 M tiles
#define NT (SDIM / 16)   // 512 N tiles
#define KC (DDIM / 32)   // 8 K chunks

// Workspace carve-up (bytes): AH | AL | BH | BL
#define AH_OFF 0
#define AL_OFF (KC * MT * 32 * 16 * 2)                   // 2 MB
#define BH_OFF (2 * (KC * MT * 32 * 16 * 2))             // 4 MB
#define BL_OFF (BH_OFF + (KC * NT * 32 * 16 * 2))        // 8 MB
#define WS_NEEDED (BH_OFF + 2 * (KC * NT * 32 * 16 * 2)) // 12 MB

__device__ __forceinline__ unsigned short bf16_rne(float x) {
  unsigned u = __float_as_uint(x);
  u += 0x7FFFu + ((u >> 16) & 1u);          // round-to-nearest-even to bf16
  return (unsigned short)(u >> 16);
}
__device__ __forceinline__ float bf16f(unsigned short h) {
  return __uint_as_float(((unsigned)h) << 16);
}
__device__ __forceinline__ void bf16_split(float x, unsigned short& h,
                                           unsigned short& l) {
  h = bf16_rne(x);
  l = bf16_rne(x - bf16f(h));
}
__device__ __forceinline__ v16bf cat8(v8bf a, v8bf b) {
  return __builtin_shufflevector(a, b, 0,1,2,3,4,5,6,7,8,9,10,11,12,13,14,15);
}
__device__ __forceinline__ v16bf ldfrag(const unsigned short* __restrict__ p) {
  return cat8(*reinterpret_cast<const v8bf*>(p),
              *reinterpret_cast<const v8bf*>(p + 8));
}

// ---------------------------------------------------------------------------
// Pre-pass A: inputs[4096x256] f32 -> hi/lo bf16 in A-fragment records.
// Record (kc, mt, lane) = 16 halves: lane r=lane&15 row of tile mt,
// halves[0..7]=K kb..kb+7, halves[8..15]=K kb+16..kb+23, kb=kc*32+(lane>>4)*8.
// 65536 records, one per thread.
// ---------------------------------------------------------------------------
__global__ __launch_bounds__(256) void prep_a_kernel(
    const float* __restrict__ inputs, unsigned short* __restrict__ AH,
    unsigned short* __restrict__ AL) {
  const int rec  = blockIdx.x * 256 + threadIdx.x;   // 0 .. 65535
  const int lane = rec & 31;
  const int mt   = (rec >> 5) & (MT - 1);
  const int kc   = rec >> 13;
  const int row  = mt * 16 + (lane & 15);
  const int kb   = kc * 32 + (lane >> 4) * 8;
  const float* src = inputs + (size_t)row * DDIM + kb;

  float f[16];
  const float4* s4 = reinterpret_cast<const float4*>(src);
  float4 t0 = s4[0], t1 = s4[1], t2 = s4[4], t3 = s4[5];  // +0,+4,+16,+20
  f[0]=t0.x; f[1]=t0.y; f[2]=t0.z; f[3]=t0.w;
  f[4]=t1.x; f[5]=t1.y; f[6]=t1.z; f[7]=t1.w;
  f[8]=t2.x; f[9]=t2.y; f[10]=t2.z; f[11]=t2.w;
  f[12]=t3.x; f[13]=t3.y; f[14]=t3.z; f[15]=t3.w;

  v8us h0, h1, l0, l1;
#pragma unroll
  for (int e = 0; e < 8; ++e) {
    unsigned short h, l;
    bf16_split(f[e], h, l);     h0[e] = h; l0[e] = l;
    bf16_split(f[e + 8], h, l); h1[e] = h; l1[e] = l;
  }
  v8us* dh = reinterpret_cast<v8us*>(AH + (size_t)rec * 16);
  v8us* dl = reinterpret_cast<v8us*>(AL + (size_t)rec * 16);
  dh[0] = h0; dh[1] = h1;
  dl[0] = l0; dl[1] = l1;
}

// ---------------------------------------------------------------------------
// Pre-pass B: gathered weight rows -> hi/lo bf16 in B-fragment records.
// Record (kc, nt, lane) = 16 halves: N = nt*16 + (lane&15), K order as in A (kb..kb+7, kb+16..kb+23).
// 131072 records, one per thread. Per j, the 32 lanes read one float each from
// 16 gathered weight rows (two lanes per row, 8 floats apart).
// ---------------------------------------------------------------------------
__global__ __launch_bounds__(256) void prep_b_kernel(
    const float* __restrict__ weight, const int* __restrict__ sample_ids,
    unsigned short* __restrict__ BH, unsigned short* __restrict__ BL) {
  const int rec  = blockIdx.x * 256 + threadIdx.x;   // 0 .. 131071
  const int lane = rec & 31;
  const int nt   = (rec >> 5) & (NT - 1);
  const int kc   = rec >> 14;
  const int k    = kc * 32 + (lane >> 4) * 8;

  v8us h0, h1, l0, l1;
#pragma unroll
  for (int j = 0; j < 16; ++j) {
    int sid = sample_ids[nt * 16 + (lane & 15)];
    float x = weight[(size_t)sid * DDIM + k + (j & 7) + (j >> 3) * 16];
    unsigned short h, l;
    bf16_split(x, h, l);
    if (j < 8) { h0[j] = h; l0[j] = l; }
    else       { h1[j - 8] = h; l1[j - 8] = l; }
  }
  v8us* dh = reinterpret_cast<v8us*>(BH + (size_t)rec * 16);
  v8us* dl = reinterpret_cast<v8us*>(BL + (size_t)rec * 16);
  dh[0] = h0; dh[1] = h1;
  dl[0] = l0; dl[1] = l1;
}

// ---------------------------------------------------------------------------
// GEMM from pre-swizzled fragments. Block = 256 threads (8 waves),
// block tile 128x128, wave tile 64(M)x32(N) = 4x2 WMMA tiles.
// Inner loop: 24 global_load_b128 (L2 hits) + 24 v_wmma per K chunk.
// No LDS, no barriers.
// ---------------------------------------------------------------------------
__global__ __launch_bounds__(256) void sampled_logits_gemm_pre(
    const unsigned short* __restrict__ AH, const unsigned short* __restrict__ AL,
    const unsigned short* __restrict__ BH, const unsigned short* __restrict__ BL,
    const int* __restrict__ labels, const int* __restrict__ sample_ids,
    const float* __restrict__ sample_freq, const float* __restrict__ bias,
    float* __restrict__ out) {
  const int tid  = threadIdx.x;
  const int lane = tid & 31;
  const int wv   = tid >> 5;
  const int wm   = wv >> 2;   // 0..1
  const int wn   = wv & 3;    // 0..3
  const int bn0  = blockIdx.x * 128;
  const int bm0  = blockIdx.y * 128;
  const int mt0  = (bm0 >> 4) + wm * 4;   // first of 4 M tiles for this wave
  const int nt0  = (bn0 >> 4) + wn * 2;   // first of 2 N tiles for this wave

  v8f acc[4][2];
#pragma unroll
  for (int mi = 0; mi < 4; ++mi)
#pragma unroll
    for (int ni = 0; ni < 2; ++ni)
#pragma unroll
      for (int i = 0; i < 8; ++i) acc[mi][ni][i] = 0.0f;

  for (int kc = 0; kc < KC; ++kc) {
    v16bf fah[4], fal[4], fbh[2], fbl[2];
#pragma unroll
    for (int mi = 0; mi < 4; ++mi) {
      size_t r = (((size_t)kc * MT + mt0 + mi) * 32 + lane) * 16;
      fah[mi] = ldfrag(AH + r);
      fal[mi] = ldfrag(AL + r);
    }
#pragma unroll
    for (int ni = 0; ni < 2; ++ni) {
      size_t r = (((size_t)kc * NT + nt0 + ni) * 32 + lane) * 16;
      fbh[ni] = ldfrag(BH + r);
      fbl[ni] = ldfrag(BL + r);
    }
#pragma unroll
    for (int mi = 0; mi < 4; ++mi)
#pragma unroll
      for (int ni = 0; ni < 2; ++ni) {
        v8f c = acc[mi][ni];
        c = __builtin_amdgcn_wmma_f32_16x16x32_bf16(false, fal[mi], false,
                                                    fbh[ni], (short)0, c,
                                                    false, false);
        c = __builtin_amdgcn_wmma_f32_16x16x32_bf16(false, fah[mi], false,
                                                    fbl[ni], (short)0, c,
                                                    false, false);
        c = __builtin_amdgcn_wmma_f32_16x16x32_bf16(false, fah[mi], false,
                                                    fbh[ni], (short)0, c,
                                                    false, false);
        acc[mi][ni] = c;
      }
  }

  // Epilogue: fused bias + (-log sample_freq), accidental-hit mask, store.
  const int el15 = lane & 15;
  const int ehi8 = (lane >> 4) * 8;
#pragma unroll
  for (int ni = 0; ni < 2; ++ni) {
    int gcol  = bn0 + wn * 32 + ni * 16 + el15;
    int sidc  = sample_ids[gcol];
    float cad = bias[sidc] - logf(sample_freq[gcol]);
#pragma unroll
    for (int mi = 0; mi < 4; ++mi) {
      int grow0 = bm0 + wm * 64 + mi * 16 + ehi8;
#pragma unroll
      for (int i = 0; i < 8; ++i) {
        int grow = grow0 + i;
        float v = acc[mi][ni][i] + cad;
        if (labels[grow] == sidc) v = -1e37f;
        out[(size_t)grow * OUTSTRIDE + 1 + gcol] = v;
      }
    }
  }
}

// ---------------------------------------------------------------------------
// Fallback fused kernel (LDS staging + in-kernel conversion), used only if
// the workspace is too small for the pre-pass path.
// ---------------------------------------------------------------------------
__global__ __launch_bounds__(256) void sampled_logits_gemm_fused(
    const float* __restrict__ inputs, const int* __restrict__ labels,
    const int* __restrict__ sample_ids, const float* __restrict__ sample_freq,
    const float* __restrict__ weight, const float* __restrict__ bias,
    float* __restrict__ out) {
  constexpr int AS = 40;
  constexpr int BS = 136;
  __shared__ __align__(16) unsigned short Ah[128 * AS];
  __shared__ __align__(16) unsigned short Al[128 * AS];
  __shared__ __align__(16) unsigned short Bh[32 * BS];
  __shared__ __align__(16) unsigned short Bl[32 * BS];
  __shared__ float col_add[128];
  __shared__ int   sid_s[128];
  __shared__ int   lab_s[128];

  const int tid  = threadIdx.x;
  const int lane = tid & 31;
  const int wv   = tid >> 5;
  const int wm   = wv >> 2;
  const int wn   = wv & 3;
  const int bn0  = blockIdx.x * 128;
  const int bm0  = blockIdx.y * 128;

  if (tid < 128) {
    int s = sample_ids[bn0 + tid];
    sid_s[tid]   = s;
    col_add[tid] = bias[s] - logf(sample_freq[bn0 + tid]);
  } else {
    int r = tid - 128;
    lab_s[r] = labels[bm0 + r];
  }
  __syncthreads();

  v8f acc[4][2];
#pragma unroll
  for (int mi = 0; mi < 4; ++mi)
#pragma unroll
    for (int ni = 0; ni < 2; ++ni)
#pragma unroll
      for (int i = 0; i < 8; ++i) acc[mi][ni][i] = 0.0f;

  const int lr   = tid >> 1;
  const int koff = (tid & 1) * 16;
  const float* arow = inputs + (size_t)(bm0 + lr) * DDIM;
  const float* brow = weight + (size_t)sid_s[lr] * DDIM;

  for (int kc = 0; kc < DDIM; kc += 32) {
    float4 ar[4], br[4];
    const float4* ap = reinterpret_cast<const float4*>(arow + kc + koff);
    const float4* bp = reinterpret_cast<const float4*>(brow + kc + koff);
#pragma unroll
    for (int j = 0; j < 4; ++j) { ar[j] = ap[j]; br[j] = bp[j]; }

    __syncthreads();
    const float* af  = reinterpret_cast<const float*>(ar);
    const float* bfp = reinterpret_cast<const float*>(br);
#pragma unroll
    for (int e = 0; e < 16; ++e) {
      int k = koff + e;
      unsigned short h, l;
      bf16_split(af[e], h, l);
      Ah[lr * AS + k] = h; Al[lr * AS + k] = l;
      bf16_split(bfp[e], h, l);
      Bh[lr * 32 + k] = h; Bl[lr * 32 + k] = l;
    }
    __syncthreads();

    const int r15  = lane & 15;
    const int hsel = (lane >> 4) * 8;
    v16bf fah[4], fal[4], fbh[2], fbl[2];
#pragma unroll
    for (int mi = 0; mi < 4; ++mi) {
      int base = (wm * 64 + mi * 16 + r15) * AS + hsel;
      fah[mi] = cat8(*(const v8bf*)&Ah[base], *(const v8bf*)&Ah[base + 16]);
      fal[mi] = cat8(*(const v8bf*)&Al[base], *(const v8bf*)&Al[base + 16]);
    }
#pragma unroll
    for (int ni = 0; ni < 2; ++ni) {
      int base = (wn * 32 + ni * 16 + r15) * 32 + hsel;
      fbh[ni] = cat8(*(const v8bf*)&Bh[base], *(const v8bf*)&Bh[base + 16]);
      fbl[ni] = cat8(*(const v8bf*)&Bl[base], *(const v8bf*)&Bl[base + 16]);
    }
#pragma unroll
    for (int mi = 0; mi < 4; ++mi)
#pragma unroll
      for (int ni = 0; ni < 2; ++ni) {
        v8f c = acc[mi][ni];
        c = __builtin_amdgcn_wmma_f32_16x16x32_bf16(false, fal[mi], false,
                                                    fbh[ni], (short)0, c,
                                                    false, false);
        c = __builtin_amdgcn_wmma_f32_16x16x32_bf16(false, fah[mi], false,
                                                    fbl[ni], (short)0, c,
                                                    false, false);
        c = __builtin_amdgcn_wmma_f32_16x16x32_bf16(false, fah[mi], false,
                                                    fbh[ni], (short)0, c,
                                                    false, false);
        acc[mi][ni] = c;
      }
  }

  const int el15 = lane & 15;
  const int ehi8 = (lane >> 4) * 8;
#pragma unroll
  for (int ni = 0; ni < 2; ++ni) {
    int cib   = wn * 32 + ni * 16 + el15;
    int sidc  = sid_s[cib];
    float cad = col_add[cib];
    size_t gcol = (size_t)(bn0 + cib);
#pragma unroll
    for (int mi = 0; mi < 4; ++mi) {
      int rb = wm * 64 + mi * 16 + ehi8;
#pragma unroll
      for (int i = 0; i < 8; ++i) {
        int rib = rb + i;
        float v = acc[mi][ni][i] + cad;
        if (lab_s[rib] == sidc) v = -1e37f;
        out[(size_t)(bm0 + rib) * OUTSTRIDE + 1 + gcol] = v;
      }
    }
  }
}

// ---------------------------------------------------------------------------
// True logits (column 0) + zero targets. One wave per row.
// ---------------------------------------------------------------------------
__global__ __launch_bounds__(256) void true_logits_kernel(
    const float* __restrict__ inputs, const int* __restrict__ labels,
    const float* __restrict__ true_freq, const float* __restrict__ weight,
    const float* __restrict__ bias, float* __restrict__ out) {
  const int lane = threadIdx.x & 31;
  const int wv   = threadIdx.x >> 5;
  const int row  = blockIdx.x * 8 + wv;
  if (row >= BDIM) return;
  const int lab = labels[row];
  const float* x = inputs + (size_t)row * DDIM;
  const float* w = weight + (size_t)lab * DDIM;
  float s = 0.0f;
#pragma unroll
  for (int k = lane; k < DDIM; k += 32) s = fmaf(x[k], w[k], s);
#pragma unroll
  for (int m = 16; m > 0; m >>= 1) s += __shfl_xor(s, m, 32);
  if (lane == 0) {
    out[(size_t)row * OUTSTRIDE] = s + bias[lab] - logf(true_freq[row]);
    out[(size_t)BDIM * OUTSTRIDE + row] = 0.0f;  // int32 0 bit pattern
  }
}

// ---------------------------------------------------------------------------
extern "C" void kernel_launch(void* const* d_in, const int* in_sizes, int n_in,
                              void* d_out, int out_size, void* d_ws,
                              size_t ws_size, hipStream_t stream) {
  const float* inputs      = (const float*)d_in[0];
  const int*   labels      = (const int*)d_in[1];
  const int*   sample_ids  = (const int*)d_in[2];
  const float* true_freq   = (const float*)d_in[3];
  const float* sample_freq = (const float*)d_in[4];
  const float* weight      = (const float*)d_in[5];
  const float* bias        = (const float*)d_in[6];
  float* out = (float*)d_out;

  dim3 grid(SDIM / 128, BDIM / 128);   // 64 x 32 tiles

  if (ws_size >= (size_t)WS_NEEDED && d_ws != nullptr) {
    unsigned short* AH = (unsigned short*)((char*)d_ws + AH_OFF);
    unsigned short* AL = (unsigned short*)((char*)d_ws + AL_OFF);
    unsigned short* BH = (unsigned short*)((char*)d_ws + BH_OFF);
    unsigned short* BL = (unsigned short*)((char*)d_ws + BL_OFF);
    prep_a_kernel<<<KC * MT * 32 / 256, 256, 0, stream>>>(inputs, AH, AL);
    prep_b_kernel<<<KC * NT * 32 / 256, 256, 0, stream>>>(weight, sample_ids,
                                                          BH, BL);
    sampled_logits_gemm_pre<<<grid, 256, 0, stream>>>(
        AH, AL, BH, BL, labels, sample_ids, sample_freq, bias, out);
  } else {
    sampled_logits_gemm_fused<<<grid, 256, 0, stream>>>(
        inputs, labels, sample_ids, sample_freq, weight, bias, out);
  }
  true_logits_kernel<<<BDIM / 8, 256, 0, stream>>>(inputs, labels, true_freq,
                                                   weight, bias, out);
}
